// GATModel_24507083391314
// MI455X (gfx1250) — matrix operand order repaired and checked by reading, not compiled, not run on hardware
//
#include <hip/hip_runtime.h>
#include <hip/hip_bf16.h>

typedef __attribute__((ext_vector_type(16))) _Float16 v16h;
typedef __attribute__((ext_vector_type(8)))  _Float16 v8h;
typedef __attribute__((ext_vector_type(8)))  float    v8f;

#define NEG_SLOPE 0.2f

// ---------- monotone float<->uint key for atomic max ----------
__device__ __forceinline__ unsigned fkey(float f) {
    unsigned b = __float_as_uint(f);
    return (b & 0x80000000u) ? ~b : (b | 0x80000000u);
}
__device__ __forceinline__ float funkey(unsigned u) {
    return (u & 0x80000000u) ? __uint_as_float(u & 0x7fffffffu)
                             : __uint_as_float(~u);
}

// ---------- f32 -> f16 row-major convert (A operands), 4 elems/thread ----------
__global__ void f32_to_f16(const float* __restrict__ in, _Float16* __restrict__ out,
                           long long n4) {
    long long i = (long long)blockIdx.x * blockDim.x + threadIdx.x;
    if (i >= n4) return;
    float4 v = ((const float4*)in)[i];
    _Float16 h0 = (_Float16)v.x, h1 = (_Float16)v.y,
             h2 = (_Float16)v.z, h3 = (_Float16)v.w;
    out[i * 4 + 0] = h0; out[i * 4 + 1] = h1;
    out[i * 4 + 2] = h2; out[i * 4 + 3] = h3;
}

// ---------- swizzle W[K x Nc] f32 into per-lane WMMA B layout (f16) ----------
// Bsw[(((tn*(K/32)) + k0s)*32 + lane)*16 + j] = W[(k0s*32 + (lane>>4)*16 + j)*Nc + tn*16 + (lane&15)]
__global__ void swizzle_w(const float* __restrict__ W, _Float16* __restrict__ Bsw,
                          int K, int Nc) {
    int idx = blockIdx.x * blockDim.x + threadIdx.x;
    if (idx >= K * Nc) return;
    int j    = idx & 15;
    int l    = (idx >> 4) & 31;
    int rest = idx >> 9;
    int ksteps = K >> 5;
    int k0s  = rest % ksteps;
    int tn   = rest / ksteps;
    int k = k0s * 32 + (l >> 4) * 16 + j;
    int n = tn * 16 + (l & 15);
    Bsw[idx] = (_Float16)W[(size_t)k * Nc + n];
}

// ---------- WMMA GEMM: C[M x Nc] = Ah[M x K] * Bsw, f16 in, f32 out ----------
// One wave per 16x16 tile; K fully unrolled -> 4x v_wmma_f32_16x16x32_f16.
template <int K>
__global__ void wmma_gemm(const _Float16* __restrict__ Ah,
                          const _Float16* __restrict__ Bsw,
                          float* __restrict__ C, int M, int Nc) {
    int wave = blockIdx.x * (blockDim.x >> 5) + (threadIdx.x >> 5);
    int lane = threadIdx.x & 31;
    int ntn = Nc >> 4;
    int total = (M >> 4) * ntn;
    if (wave >= total) return;                 // wave-uniform: EXEC stays all-ones
    int tm = wave / ntn, tn = wave % ntn;
    int row_base = tm << 4;
    int m = lane & 15, half = lane >> 4;
    constexpr int KS = K >> 5;

    const _Float16* Arow = Ah + (size_t)(row_base + m) * K + half * 16;
    const _Float16* Bp   = Bsw + ((size_t)(tn * KS) * 32 + lane) * 16;

    v8f acc = {};
#pragma unroll
    for (int ks = 0; ks < KS; ++ks) {
        v16h a, b;
        // A: elements 0..15 -> K = k0+half*16+j
        *(v8h*)&a       = *(const v8h*)(Arow + ks * 32);
        *((v8h*)&a + 1) = *(const v8h*)(Arow + ks * 32 + 8);
        b = *(const v16h*)(Bp + (size_t)ks * 512);
        acc = __builtin_amdgcn_wmma_f32_16x16x32_f16(
            false, a, false, b, (short)0, acc, false, false);
    }
    int col = (tn << 4) + m;
#pragma unroll
    for (int r = 0; r < 8; ++r)
        C[(size_t)(row_base + r + 8 * half) * Nc + col] = acc[r];
}

// ---------- materialize edge list (+self loops) as int2 ----------
__global__ void build_edges(const long long* __restrict__ ei, long long E,
                            long long Etot, int2* __restrict__ e32) {
    long long e = (long long)blockIdx.x * blockDim.x + threadIdx.x;
    if (e >= Etot) return;
    if (e < E) e32[e] = make_int2((int)ei[e], (int)ei[E + e]);
    else { int v = (int)(e - E); e32[e] = make_int2(v, v); }
}

// ---------- per-node attention dots: a_s[n,h] = <feat[n,h,:], att_src[h,:]> ----------
template <int H, int C>
__global__ void att_scores(const float* __restrict__ feat,
                           const float* __restrict__ att_src,
                           const float* __restrict__ att_dst,
                           float* __restrict__ as, float* __restrict__ ad, int N) {
    int n = blockIdx.x * blockDim.x + threadIdx.x;
    if (n >= N) return;
    const float* fp = feat + (size_t)n * H * C;
#pragma unroll
    for (int hh = 0; hh < H; ++hh) {
        float s = 0.f, d = 0.f;
#pragma unroll
        for (int c = 0; c < C; ++c) {
            float v = fp[hh * C + c];
            s += v * att_src[hh * C + c];
            d += v * att_dst[hh * C + c];
        }
        as[(size_t)n * H + hh] = s;
        ad[(size_t)n * H + hh] = d;
    }
}

// ---------- pass 1: segment max over dst of leaky_relu(a_s[src]+a_d[dst]) ----------
template <int H>
__global__ void edge_max(const int2* __restrict__ e32, long long Etot,
                         const float* __restrict__ as, const float* __restrict__ ad,
                         unsigned* __restrict__ mkey) {
    long long idx = (long long)blockIdx.x * blockDim.x + threadIdx.x;
    long long e = idx / H;
    int hh = (int)(idx % H);
    if (e >= Etot) return;
    int2 sd = e32[e];
    float v = as[(size_t)sd.x * H + hh] + ad[(size_t)sd.y * H + hh];
    v = v > 0.f ? v : NEG_SLOPE * v;
    atomicMax(&mkey[(size_t)sd.y * H + hh], fkey(v));
}

// ---------- pass 2: denom[dst,h] += exp(e - m[dst,h]) ----------
template <int H>
__global__ void edge_denom(const int2* __restrict__ e32, long long Etot,
                           const float* __restrict__ as, const float* __restrict__ ad,
                           const unsigned* __restrict__ mkey, float* __restrict__ denom) {
    long long idx = (long long)blockIdx.x * blockDim.x + threadIdx.x;
    long long e = idx / H;
    int hh = (int)(idx % H);
    if (e >= Etot) return;
    int2 sd = e32[e];
    float v = as[(size_t)sd.x * H + hh] + ad[(size_t)sd.y * H + hh];
    v = v > 0.f ? v : NEG_SLOPE * v;
    float mv = funkey(mkey[(size_t)sd.y * H + hh]);
    atomicAdd(&denom[(size_t)sd.y * H + hh], __expf(v - mv));
}

// ---------- pass 3: out[dst,:] += alpha[h] * feat[src,:] ----------
// 4 channels/thread (float4 gather + 4 f32 atomics); alpha broadcast via LDS.
template <int H, int C>
__global__ void edge_scatter(const int2* __restrict__ e32, long long Etot,
                             const float* __restrict__ as, const float* __restrict__ ad,
                             const unsigned* __restrict__ mkey,
                             const float* __restrict__ denom,
                             const float* __restrict__ feat, float* __restrict__ out) {
    constexpr int W   = H * C;
    constexpr int TPE = W / 4;          // threads per edge
    constexpr int EPB = 256 / TPE;      // edges per block
    __shared__ float alpha_s[EPB][H];
    int el = threadIdx.x / TPE;
    int w4 = threadIdx.x % TPE;
    long long e = (long long)blockIdx.x * EPB + el;
    bool valid = e < Etot;
    int2 sd = valid ? e32[e] : make_int2(0, 0);
    const float4* src_row = (const float4*)(feat + (size_t)sd.x * W);
    if (valid) __builtin_prefetch(&src_row[w4], 0, 0);   // global_prefetch_b8
    if (valid && w4 < H) {
        int hh = w4;
        float v = as[(size_t)sd.x * H + hh] + ad[(size_t)sd.y * H + hh];
        v = v > 0.f ? v : NEG_SLOPE * v;
        float mv = funkey(mkey[(size_t)sd.y * H + hh]);
        alpha_s[el][hh] = __expf(v - mv) / denom[(size_t)sd.y * H + hh];
    }
    __syncthreads();
    if (!valid) return;
    float alpha = alpha_s[el][(w4 * 4) / C];
    float4 f = src_row[w4];
    float* op = out + (size_t)sd.y * W + w4 * 4;
    atomicAdd(op + 0, alpha * f.x);
    atomicAdd(op + 1, alpha * f.y);
    atomicAdd(op + 2, alpha * f.z);
    atomicAdd(op + 3, alpha * f.w);
}

// ---------- bias + ELU (in place, float4) ----------
__global__ void bias_elu4(float4* __restrict__ x, const float* __restrict__ b,
                          long long n4, int W4) {
    long long i = (long long)blockIdx.x * blockDim.x + threadIdx.x;
    if (i >= n4) return;
    int g = (int)(i % W4) * 4;
    float4 v = x[i];
    v.x += b[g + 0]; v.y += b[g + 1]; v.z += b[g + 2]; v.w += b[g + 3];
    v.x = v.x > 0.f ? v.x : __expf(v.x) - 1.f;
    v.y = v.y > 0.f ? v.y : __expf(v.y) - 1.f;
    v.z = v.z > 0.f ? v.z : __expf(v.z) - 1.f;
    v.w = v.w > 0.f ? v.w : __expf(v.w) - 1.f;
    x[i] = v;
}

// ---------- final: out[n,:] = log_softmax(out[n,:] + b2) over 32 classes ----------
__global__ void final_logsoftmax(float* __restrict__ out, const float* __restrict__ b2, int N) {
    int n = blockIdx.x * blockDim.x + threadIdx.x;
    if (n >= N) return;
    float v[32];
    float mx = -1e30f;
#pragma unroll
    for (int c = 0; c < 32; ++c) {
        v[c] = out[(size_t)n * 32 + c] + b2[c];
        mx = v[c] > mx ? v[c] : mx;
    }
    float s = 0.f;
#pragma unroll
    for (int c = 0; c < 32; ++c) s += __expf(v[c] - mx);
    float ls = logf(s);
#pragma unroll
    for (int c = 0; c < 32; ++c) out[(size_t)n * 32 + c] = v[c] - mx - ls;
}

extern "C" void kernel_launch(void* const* d_in, const int* in_sizes, int n_in,
                              void* d_out, int out_size, void* d_ws, size_t ws_size,
                              hipStream_t stream) {
    const float*     x    = (const float*)d_in[0];
    const long long* ei   = (const long long*)d_in[1];   // int64 edge_index [2,E]
    const float*     W1   = (const float*)d_in[2];
    const float*     asr1 = (const float*)d_in[3];
    const float*     adt1 = (const float*)d_in[4];
    const float*     b1   = (const float*)d_in[5];
    const float*     W2   = (const float*)d_in[6];
    const float*     asr2 = (const float*)d_in[7];
    const float*     adt2 = (const float*)d_in[8];
    const float*     b2   = (const float*)d_in[9];
    float* out = (float*)d_out;

    const int IN_C = 128, HID = 32, HEADS = 4, OUT_C = 32;
    const int W1cols = HEADS * HID;                      // 128
    const long long N    = in_sizes[0] / IN_C;           // 100000
    const long long E    = in_sizes[1] / 2;              // 1600000
    const long long Etot = E + N;                        // + self loops

    // ---- workspace carve-up ----
    char* p = (char*)d_ws;
    auto take = [&](size_t bytes) { void* q = p; p += ((bytes + 255) & ~(size_t)255); return q; };
    _Float16* Ah   = (_Float16*)take((size_t)N * 128 * 2);  // x_f16, later reused for elu(agg1)_f16
    float*    h1   = (float*)take((size_t)N * 128 * 4);     // layer1 features; reused as h2
    float*    as1  = (float*)take((size_t)N * 4 * 4);
    float*    ad1  = (float*)take((size_t)N * 4 * 4);
    unsigned* mk1  = (unsigned*)take((size_t)N * 4 * 4);
    float*    den1 = (float*)take((size_t)N * 4 * 4);
    float*    agg1 = (float*)take((size_t)N * 128 * 4);
    float*    as2  = (float*)take((size_t)N * 4);
    float*    ad2  = (float*)take((size_t)N * 4);
    unsigned* mk2  = (unsigned*)take((size_t)N * 4);
    float*    den2 = (float*)take((size_t)N * 4);
    _Float16* Bsw1 = (_Float16*)take((size_t)IN_C * W1cols * 2);
    _Float16* Bsw2 = (_Float16*)take((size_t)W1cols * OUT_C * 2);
    int2*     e32  = (int2*)take((size_t)Etot * 8);
    float*    h2   = h1;                                    // alias: h1 dead before GEMM2
    (void)ws_size;

    // ---- zero-init accumulators every call (deterministic, capture-safe) ----
    hipMemsetAsync(mk1,  0, (size_t)N * 4 * 4,   stream);   // key 0 == below -inf sentinel
    hipMemsetAsync(den1, 0, (size_t)N * 4 * 4,   stream);
    hipMemsetAsync(agg1, 0, (size_t)N * 128 * 4, stream);
    hipMemsetAsync(mk2,  0, (size_t)N * 4,       stream);
    hipMemsetAsync(den2, 0, (size_t)N * 4,       stream);
    hipMemsetAsync(out,  0, (size_t)N * 32 * 4,  stream);

    // ---- one-time prep: edges, f16 operands, swizzled weights ----
    build_edges<<<(int)((Etot + 255) / 256), 256, 0, stream>>>(ei, E, Etot, e32);
    {   long long n4 = N * 128 / 4;
        f32_to_f16<<<(int)((n4 + 255) / 256), 256, 0, stream>>>(x, Ah, n4);
    }
    swizzle_w<<<(IN_C * W1cols + 255) / 256, 256, 0, stream>>>(W1, Bsw1, IN_C, W1cols);
    swizzle_w<<<(W1cols * OUT_C + 255) / 256, 256, 0, stream>>>(W2, Bsw2, W1cols, OUT_C);

    // ================= layer 1 =================
    {   // h1 = x @ W1   (100000x128 * 128x128)
        int tiles = (int)(N / 16) * (W1cols / 16);
        wmma_gemm<128><<<(tiles + 7) / 8, 256, 0, stream>>>(Ah, Bsw1, h1, (int)N, W1cols);
    }
    att_scores<4, 32><<<(int)((N + 255) / 256), 256, 0, stream>>>(h1, asr1, adt1, as1, ad1, (int)N);

    {   long long th = Etot * 4;
        edge_max<4><<<(int)((th + 255) / 256), 256, 0, stream>>>(e32, Etot, as1, ad1, mk1);
        edge_denom<4><<<(int)((th + 255) / 256), 256, 0, stream>>>(e32, Etot, as1, ad1, mk1, den1);
    }
    {   // 8 edges per 256-thread block (32 threads/edge, float4 channels)
        long long blocks = (Etot + 7) / 8;
        edge_scatter<4, 32><<<(int)blocks, 256, 0, stream>>>(e32, Etot, as1, ad1, mk1, den1, h1, agg1);
    }
    {   long long n4 = N * 128 / 4;
        bias_elu4<<<(int)((n4 + 255) / 256), 256, 0, stream>>>((float4*)agg1, b1, n4, 32);
    }

    // ================= layer 2 =================
    {   long long n4 = N * 128 / 4;   // convert elu(agg1) to f16 (reuse Ah)
        f32_to_f16<<<(int)((n4 + 255) / 256), 256, 0, stream>>>(agg1, Ah, n4);
    }
    {   // h2 = elu(agg1) @ W2   (100000x128 * 128x32)
        int tiles = (int)(N / 16) * (OUT_C / 16);
        wmma_gemm<128><<<(tiles + 7) / 8, 256, 0, stream>>>(Ah, Bsw2, h2, (int)N, OUT_C);
    }
    att_scores<1, 32><<<(int)((N + 255) / 256), 256, 0, stream>>>(h2, asr2, adt2, as2, ad2, (int)N);

    {   long long th = Etot;
        edge_max<1><<<(int)((th + 255) / 256), 256, 0, stream>>>(e32, Etot, as2, ad2, mk2);
        edge_denom<1><<<(int)((th + 255) / 256), 256, 0, stream>>>(e32, Etot, as2, ad2, mk2, den2);
    }
    {   // 32 edges per 256-thread block (8 threads/edge), accumulate into d_out
        long long blocks = (Etot + 31) / 32;
        edge_scatter<1, 32><<<(int)blocks, 256, 0, stream>>>(e32, Etot, as2, ad2, mk2, den2, h2, out);
    }
    final_logsoftmax<<<(int)((N + 255) / 256), 256, 0, stream>>>(out, b2, (int)N);
}
